// EdgePropertyPredictionModel_43361989820686
// MI455X (gfx1250) — matrix operand order repaired and checked by reading, not compiled, not run on hardware
//
#include <hip/hip_runtime.h>
#include <hip/hip_bf16.h>

typedef __attribute__((ext_vector_type(16))) __bf16 bf16x16;
typedef __attribute__((ext_vector_type(8)))  float  f32x8;

#define LRELU_SLOPE 0.2f
#define EPSV 1e-5f

static inline int cdiv(int a, int b) { return (a + b - 1) / b; }

// ---------------------------------------------------------------------------
// WMMA GEMM: C[M,Ncols] = A[M,Kp] (row stride lda, f32) @ Wb[Ncols,Kp]^T (bf16)
// Kp is a multiple of 32 (weights/activations pre-padded with zeros).
// The 16xKp A tile (shared by all 4 waves of the block) is staged into LDS
// once via GLOBAL_LOAD_ASYNC_TO_LDS_B128 (ASYNCcnt), then each wave computes
// a 16x64 output tile: one A fragment (from LDS) feeds 4 WMMAs.
// ---------------------------------------------------------------------------
__global__ void __launch_bounds__(128)
wmma_gemm_kernel(const float* __restrict__ A, int lda,
                 const __bf16* __restrict__ Wb, int Kp,
                 const float* __restrict__ bias,
                 float* __restrict__ C, int ldc,
                 int M, int Ncols, int relu)
{
    __shared__ float tileA[16 * 256];          // 16 rows x Kp<=256, 16 KB max

    const int tid  = threadIdx.x;
    const int lane = tid & 31;
    const int wave = tid >> 5;
    const int m0   = blockIdx.x * 16;
    const int l15  = lane & 15;
    const int half = lane >> 4;

    // ---- Async-stage A tile to LDS (whole block cooperates) ----
    {
        const int qperrow = Kp >> 2;           // float4 chunks per row
        const int nchunk  = 16 * qperrow;
        for (int c = tid; c < nchunk; c += 128) {
            int row = c / qperrow;
            int kq  = (c - row * qperrow) * 4;
            int mr  = m0 + row;
            const float* gsrc = A + (size_t)(mr < M ? mr : 0) * lda + kq;
            unsigned ldst = (unsigned)(uintptr_t)(tileA + row * Kp + kq);
            asm volatile("global_load_async_to_lds_b128 %0, %1, off"
                         :: "v"(ldst), "v"(gsrc) : "memory");
        }
        asm volatile("s_wait_asynccnt 0x0" ::: "memory");
        __syncthreads();
    }

    const int n0 = (blockIdx.y * 4 + wave) * 64;
    if (n0 >= Ncols) return;                   // wave-uniform (after barrier)

    const __bf16* Wr[4];
#pragma unroll
    for (int t = 0; t < 4; ++t) {
        int nc = n0 + t * 16 + l15;
        Wr[t] = Wb + (size_t)(nc < Ncols ? nc : 0) * Kp;
    }

    f32x8 acc[4] = {};

    const int ka = half * 16;   // A: lanes 0-15 carry K 0-15, lanes 16-31 K 16-31
    const int kb = half * 16;   // B: lanes 0-15 carry K 0-15, lanes 16-31 K 16-31
    const float* Al = tileA + l15 * Kp;

    for (int k0 = 0; k0 < Kp; k0 += 32) {
        const float4 a0 = *(const float4*)(Al + k0 + ka);
        const float4 a1 = *(const float4*)(Al + k0 + ka + 4);
        const float4 a2 = *(const float4*)(Al + k0 + ka + 8);
        const float4 a3 = *(const float4*)(Al + k0 + ka + 12);
        bf16x16 a;
        a[0]  = (__bf16)a0.x; a[1]  = (__bf16)a0.y; a[2]  = (__bf16)a0.z; a[3]  = (__bf16)a0.w;
        a[4]  = (__bf16)a1.x; a[5]  = (__bf16)a1.y; a[6]  = (__bf16)a1.z; a[7]  = (__bf16)a1.w;
        a[8]  = (__bf16)a2.x; a[9]  = (__bf16)a2.y; a[10] = (__bf16)a2.z; a[11] = (__bf16)a2.w;
        a[12] = (__bf16)a3.x; a[13] = (__bf16)a3.y; a[14] = (__bf16)a3.z; a[15] = (__bf16)a3.w;
#pragma unroll
        for (int t = 0; t < 4; ++t) {
            bf16x16 b = *(const bf16x16*)(Wr[t] + k0 + kb);
            acc[t] = __builtin_amdgcn_wmma_f32_16x16x32_bf16(false, a, false, b,
                                                             (short)0, acc[t], false, false);
        }
    }

    // C/D layout: lane l -> column (l&15); VGPR j -> row m0+j (+8 for upper half)
#pragma unroll
    for (int t = 0; t < 4; ++t) {
        const int n = n0 + t * 16 + l15;
        if (n < Ncols) {
            const float bv = bias ? bias[n] : 0.f;
#pragma unroll
            for (int j = 0; j < 8; ++j) {
                int m = m0 + half * 8 + j;
                if (m < M) {
                    float v = acc[t][j] + bv;
                    if (relu) v = fmaxf(v, 0.f);
                    C[(size_t)m * ldc + n] = v;
                }
            }
        }
    }
}

// f32 weights [Ncols,K] -> bf16 [Ncols,Kp], zero-padded K..Kp
__global__ void convert_w_kernel(const float* __restrict__ W, __bf16* __restrict__ Wb,
                                 int Ncols, int K, int Kp)
{
    long i = (long)blockIdx.x * blockDim.x + threadIdx.x;
    if (i >= (long)Ncols * Kp) return;
    int n = (int)(i / Kp);
    int k = (int)(i % Kp);
    Wb[i] = (__bf16)(k < K ? W[(size_t)n * K + k] : 0.f);
}

// zero columns 218..223 of the ld=224 FF mid buffer
__global__ void zero_pad_kernel(float* __restrict__ mid, long M)
{
    long i = (long)blockIdx.x * blockDim.x + threadIdx.x;
    if (i >= M * 6) return;
    long r = i / 6;
    int  c = 218 + (int)(i % 6);
    mid[r * 224 + c] = 0.f;
}

// ---------------------------------------------------------------------------
// Elementwise / fill helpers
// ---------------------------------------------------------------------------
__global__ void fill_f32_kernel(float* p, float v, long n)
{
    long i = (long)blockIdx.x * blockDim.x + threadIdx.x;
    if (i < n) p[i] = v;
}

__global__ void fill_u32_kernel(unsigned* p, unsigned v, long n)
{
    long i = (long)blockIdx.x * blockDim.x + threadIdx.x;
    if (i < n) p[i] = v;
}

__global__ void fill_rows_kernel(float* __restrict__ out, const float* __restrict__ bias,
                                 long M, int C)
{
    long i = (long)blockIdx.x * blockDim.x + threadIdx.x;
    if (i < M * (long)C) out[i] = bias[i % C];
}

// ---------------------------------------------------------------------------
// Attention scores: el/er[node,h] = sum_d feat[node,h,d] * a{l,r}[h,d]
// ---------------------------------------------------------------------------
__global__ void attn_score_kernel(const float* __restrict__ feat,
                                  const float* __restrict__ al,
                                  const float* __restrict__ ar,
                                  float* __restrict__ el, float* __restrict__ er, int n)
{
    int i = blockIdx.x * blockDim.x + threadIdx.x;
    if (i >= n * 2) return;
    int node = i >> 1, h = i & 1;
    const float* f = feat + (size_t)node * 128 + h * 64;
    const float* a = al + h * 64;
    const float* r = ar + h * 64;
    float sl = 0.f, sr = 0.f;
#pragma unroll 8
    for (int d = 0; d < 64; ++d) { float v = f[d]; sl += v * a[d]; sr += v * r[d]; }
    el[i] = sl; er[i] = sr;
}

// Monotone float <-> uint encoding so segment-max can use integer atomicMax
__device__ __forceinline__ unsigned fenc(float f)
{
    unsigned u = __float_as_uint(f);
    return (u & 0x80000000u) ? ~u : (u | 0x80000000u);
}
__device__ __forceinline__ float fdec(unsigned u)
{
    unsigned v = (u & 0x80000000u) ? (u ^ 0x80000000u) : ~u;
    return __uint_as_float(v);
}
__device__ __forceinline__ float lrelu(float v) { return v > 0.f ? v : LRELU_SLOPE * v; }

__global__ void edge_max_kernel(const int* __restrict__ src, const int* __restrict__ dst,
                                const float* __restrict__ el, const float* __restrict__ er,
                                unsigned* __restrict__ menc, int E)
{
    int i = blockIdx.x * blockDim.x + threadIdx.x;
    if (i >= E * 2) return;
    int e = i >> 1, h = i & 1;
    int s = src[e], d = dst[e];
    float v = lrelu(el[s * 2 + h] + er[d * 2 + h]);
    atomicMax(menc + d * 2 + h, fenc(v));
}

__global__ void edge_exp_kernel(const int* __restrict__ src, const int* __restrict__ dst,
                                const float* __restrict__ el, const float* __restrict__ er,
                                const unsigned* __restrict__ menc,
                                float* __restrict__ ex, float* __restrict__ den, int E)
{
    int i = blockIdx.x * blockDim.x + threadIdx.x;
    if (i >= E * 2) return;
    int e = i >> 1, h = i & 1;
    int s = src[e], d = dst[e];
    float v = lrelu(el[s * 2 + h] + er[d * 2 + h]);
    float x = __expf(v - fdec(menc[d * 2 + h]));
    ex[i] = x;
    atomicAdd(den + d * 2 + h, x);
}

// 32 threads per edge, each moves a float4 of the 128-wide feature row
__global__ void edge_aggr_kernel(const int* __restrict__ src, const int* __restrict__ dst,
                                 const float* __restrict__ ex, const float* __restrict__ den,
                                 const float* __restrict__ feat, float* __restrict__ out, int E)
{
    long long t = (long long)blockIdx.x * blockDim.x + threadIdx.x;
    long long e = t >> 5;
    if (e >= E) return;
    int part = (int)(t & 31);
    int c0   = part * 4;
    int h    = c0 >> 6;
    int s = src[e], d = dst[e];
    float alpha = ex[e * 2 + h] / den[d * 2 + h];
    const float4 f = *(const float4*)(feat + (size_t)s * 128 + c0);
    float* o = out + (size_t)d * 128 + c0;
    atomicAdd(o + 0, alpha * f.x);
    atomicAdd(o + 1, alpha * f.y);
    atomicAdd(o + 2, alpha * f.z);
    atomicAdd(o + 3, alpha * f.w);
}

// ---------------------------------------------------------------------------
// BatchNorm (training-mode, biased variance): one block per column
// ---------------------------------------------------------------------------
__global__ void __launch_bounds__(256)
bn_stats_kernel(const float* __restrict__ X, int ldx, int M,
                const float* __restrict__ g, const float* __restrict__ b,
                float* __restrict__ scale, float* __restrict__ shift)
{
    __shared__ float ssum[256];
    __shared__ float ssq[256];
    int c = blockIdx.x;
    float s = 0.f, q = 0.f;
    for (int r = threadIdx.x; r < M; r += 256) {
        float v = X[(size_t)r * ldx + c];
        s += v; q += v * v;
    }
    ssum[threadIdx.x] = s; ssq[threadIdx.x] = q;
    __syncthreads();
    for (int o = 128; o > 0; o >>= 1) {
        if (threadIdx.x < o) {
            ssum[threadIdx.x] += ssum[threadIdx.x + o];
            ssq[threadIdx.x]  += ssq[threadIdx.x + o];
        }
        __syncthreads();
    }
    if (threadIdx.x == 0) {
        float mu  = ssum[0] / M;
        float var = ssq[0] / M - mu * mu;
        float sc  = g[c] * rsqrtf(var + EPSV);
        scale[c] = sc;
        shift[c] = b[c] - mu * sc;
    }
}

__global__ void bn_apply_kernel(const float* __restrict__ X, int ldx,
                                float* __restrict__ Y, int ldy,
                                const float* __restrict__ scale,
                                const float* __restrict__ shift,
                                long M, int C, int relu)
{
    long i = (long)blockIdx.x * blockDim.x + threadIdx.x;
    if (i >= M * (long)C) return;
    long r = i / C; int c = (int)(i % C);
    float v = X[r * ldx + c] * scale[c] + shift[c];
    if (relu) v = fmaxf(v, 0.f);
    Y[r * ldy + c] = v;
}

__global__ void final_dot_kernel(const float* __restrict__ H, const float* __restrict__ w,
                                 float* __restrict__ out, int M)
{
    int r = blockIdx.x * blockDim.x + threadIdx.x;
    if (r >= M) return;
    const float* h = H + (size_t)r * 64;
    float s = 0.f;
#pragma unroll
    for (int d = 0; d < 64; ++d) s += h[d] * w[d];
    out[r] = s;
}

// ---------------------------------------------------------------------------
// Host-side orchestration
// ---------------------------------------------------------------------------
static void gemm(const float* A, int lda, const __bf16* Wb, int Kp, const float* bias,
                 float* C, int ldc, int M, int Ncols, int relu, hipStream_t s)
{
    dim3 block(128);
    dim3 grid(cdiv(M, 16), cdiv(Ncols, 256));   // 4 waves x 64 cols per block
    wmma_gemm_kernel<<<grid, block, 0, s>>>(A, lda, Wb, Kp, bias, C, ldc, M, Ncols, relu);
}

static void convw(const float* W, __bf16* Wb, int Ncols, int K, int Kp, hipStream_t s)
{
    long n = (long)Ncols * Kp;
    convert_w_kernel<<<(unsigned)((n + 255) / 256), 256, 0, s>>>(W, Wb, Ncols, K, Kp);
}

extern "C" void kernel_launch(void* const* d_in, const int* in_sizes, int n_in,
                              void* d_out, int out_size, void* d_ws, size_t ws_size,
                              hipStream_t stream)
{
    (void)n_in; (void)out_size; (void)ws_size;
    const float* x      = (const float*)d_in[0];
    const int*   src    = (const int*)d_in[1];
    const int*   dst    = (const int*)d_in[2];
    const float* emb_w  = (const float*)d_in[3];
    const float* emb_b  = (const float*)d_in[4];
    const float* fc_w   = (const float*)d_in[5];
    const float* attn_l = (const float*)d_in[6];
    const float* attn_r = (const float*)d_in[7];
    const float* gat_b  = (const float*)d_in[8];
    const float* bn1_g  = (const float*)d_in[9];
    const float* bn1_b  = (const float*)d_in[10];
    const float* ff_w1  = (const float*)d_in[11];
    const float* ff_b1  = (const float*)d_in[12];
    const float* ff_w2  = (const float*)d_in[13];
    const float* ff_b2  = (const float*)d_in[14];
    const float* bn2_g  = (const float*)d_in[15];
    const float* bn2_b  = (const float*)d_in[16];
    const float* mlp_w1 = (const float*)d_in[17];
    const float* mlp_bn_g = (const float*)d_in[18];
    const float* mlp_bn_b = (const float*)d_in[19];
    const float* mlp_w2 = (const float*)d_in[20];

    const int N = in_sizes[0] / 64;
    const int E = in_sizes[1];

    // Workspace carve-up. cat[N,256] doubles as hidden-state storage: layer
    // inputs are read as strided slices and BN writes the next slice.
    float* ws   = (float*)d_ws;
    float* cat  = ws;                           // N*256
    float* feat = cat  + (size_t)N * 256;       // N*128
    float* gout = feat + (size_t)N * 128;       // N*128 (reused as [N,64] later)
    float* mid  = gout + (size_t)N * 128;       // N*224 (ld=224, K-padded FF mid)
    float* el   = mid  + (size_t)N * 224;       // N*2
    float* er   = el   + (size_t)N * 2;         // N*2
    float* den  = er   + (size_t)N * 2;         // N*2
    unsigned* menc = (unsigned*)(den + (size_t)N * 2); // N*2
    float* ex   = (float*)(menc + (size_t)N * 2);      // E*2
    float* bnsc = ex + (size_t)E * 2;           // 256
    float* bnsh = bnsc + 256;                   // 256

    // bf16 weight staging (aligned to 64B)
    uintptr_t wp = (uintptr_t)(bnsh + 256);
    wp = (wp + 63) & ~(uintptr_t)63;
    __bf16* emb_wb  = (__bf16*)wp;                  // 64*64
    __bf16* fc_wb   = emb_wb  + 64 * 64;            // 3*128*64
    __bf16* ff1_wb  = fc_wb   + 3 * 128 * 64;       // 3*218*128
    __bf16* ff2_wb  = ff1_wb  + 3 * 218 * 128;      // 3*64*224
    __bf16* mlp1_wb = ff2_wb  + 3 * 64 * 224;       // 64*256

    const int TB = 256;
    const long nh   = (long)N * 2;
    const long eh   = (long)E * 2;
    const long eagg = (long)E * 32;

    // Stage all weights as bf16 (K padded to multiples of 32)
    convw(emb_w,  emb_wb,  64,      64,  64,  stream);
    convw(fc_w,   fc_wb,   3 * 128, 64,  64,  stream);
    convw(ff_w1,  ff1_wb,  3 * 218, 128, 128, stream);
    convw(ff_w2,  ff2_wb,  3 * 64,  218, 224, stream);
    convw(mlp_w1, mlp1_wb, 64,      256, 256, stream);

    // Embedding: cat[:, 0:64] = x @ emb_w^T + emb_b
    gemm(x, 64, emb_wb, 64, emb_b, cat, 256, N, 64, 0, stream);

    for (int l = 0; l < 3; ++l) {
        const float* hin = cat + (size_t)l * 64;               // [N,64], lda=256

        // feat = h @ fc_w[l]^T  -> [N,128]
        gemm(hin, 256, fc_wb + (size_t)l * 128 * 64, 64, nullptr, feat, 128, N, 128, 0, stream);

        attn_score_kernel<<<(unsigned)((nh + TB - 1) / TB), TB, 0, stream>>>(
            feat, attn_l + (size_t)l * 128, attn_r + (size_t)l * 128, el, er, N);

        fill_u32_kernel<<<(unsigned)((nh + TB - 1) / TB), TB, 0, stream>>>(menc, 0u, nh);
        fill_f32_kernel<<<(unsigned)((nh + TB - 1) / TB), TB, 0, stream>>>(den, 0.f, nh);
        fill_rows_kernel<<<(unsigned)(((long)N * 128 + TB - 1) / TB), TB, 0, stream>>>(
            gout, gat_b + (size_t)l * 128, N, 128);

        edge_max_kernel<<<(unsigned)((eh + TB - 1) / TB), TB, 0, stream>>>(
            src, dst, el, er, menc, E);
        edge_exp_kernel<<<(unsigned)((eh + TB - 1) / TB), TB, 0, stream>>>(
            src, dst, el, er, menc, ex, den, E);
        edge_aggr_kernel<<<(unsigned)((eagg + TB - 1) / TB), TB, 0, stream>>>(
            src, dst, ex, den, feat, gout, E);

        // BN1 over [N,128], in place
        bn_stats_kernel<<<128, 256, 0, stream>>>(gout, 128, N,
            bn1_g + (size_t)l * 128, bn1_b + (size_t)l * 128, bnsc, bnsh);
        bn_apply_kernel<<<(unsigned)(((long)N * 128 + TB - 1) / TB), TB, 0, stream>>>(
            gout, 128, gout, 128, bnsc, bnsh, N, 128, 0);

        // FF: relu(g @ ff_w1^T + b1) @ ff_w2^T + b2   (mid has ld=224, K-padded)
        gemm(gout, 128, ff1_wb + (size_t)l * 218 * 128, 128, ff_b1 + (size_t)l * 218,
             mid, 224, N, 218, 1, stream);
        zero_pad_kernel<<<(unsigned)(((long)N * 6 + TB - 1) / TB), TB, 0, stream>>>(mid, N);
        gemm(mid, 224, ff2_wb + (size_t)l * 64 * 224, 224, ff_b2 + (size_t)l * 64,
             gout, 64, N, 64, 0, stream);

        // BN2 -> writes the next hidden state straight into the concat slice
        bn_stats_kernel<<<64, 256, 0, stream>>>(gout, 64, N,
            bn2_g + (size_t)l * 64, bn2_b + (size_t)l * 64, bnsc, bnsh);
        bn_apply_kernel<<<(unsigned)(((long)N * 64 + TB - 1) / TB), TB, 0, stream>>>(
            gout, 64, cat + (size_t)(l + 1) * 64, 256, bnsc, bnsh, N, 64, 0);
    }

    // MLP head: relu(BN(cat @ mlp_w1^T)) @ mlp_w2^T
    gemm(cat, 256, mlp1_wb, 256, nullptr, mid, 64, N, 64, 0, stream);
    bn_stats_kernel<<<64, 256, 0, stream>>>(mid, 64, N, mlp_bn_g, mlp_bn_b, bnsc, bnsh);
    bn_apply_kernel<<<(unsigned)(((long)N * 64 + TB - 1) / TB), TB, 0, stream>>>(
        mid, 64, gout, 64, bnsc, bnsh, N, 64, 1);
    final_dot_kernel<<<(unsigned)((N + TB - 1) / TB), TB, 0, stream>>>(
        gout, mlp_w2, (float*)d_out, N);
}
